// TMPAlign_91147795956078
// MI455X (gfx1250) — compile-verified, matrix operand order checked by reading, not run on hardware
//
#include <hip/hip_runtime.h>
#include <stdint.h>

// ---------------- geometry (fixed by the reference) ----------------
#define HH 96
#define WW 96
#define PP (HH * WW)   // 9216 pixels per frame
#define TT 4

// K-chunk staged in LDS per iteration (all conv K = Cin*9 are multiples of 288)
#define KC 288
#define LDSROW (KC + 8)   // +16B pad: row stride 148 dwords -> 16 rows / 16 banks

typedef __attribute__((ext_vector_type(16))) __bf16 v16bf;
typedef __attribute__((ext_vector_type(8)))  float  v8f;

union BfFrag { v16bf v; uint4 q[2]; };

enum { ACT_NONE = 0, ACT_RELU = 1, ACT_LEAKY = 2 };

__device__ __forceinline__ unsigned short f32_to_bf16(float f) {
  unsigned int b = __float_as_uint(f);
  b += 0x7FFFu + ((b >> 16) & 1u);   // round-to-nearest-even
  return (unsigned short)(b >> 16);
}

__device__ __forceinline__ int iclampd(int v, int lo, int hi) {
  return v < lo ? lo : (v > hi ? hi : v);
}

// ---------------------------------------------------------------------------
// Implicit-GEMM 3x3 conv:  C[N x M] = act(A[M x K] * Bt[N x K]^T + bias) + res
//   A  : im2col activations, bf16, row-major (K = Cin*9 contiguous)
//   Bt : weights, bf16, row-major [Cout x K] (natural OIHW flattening)
//   C  : fp32 NCHW (channel-major, pixel contiguous)
// Block = 64M x 64N, 8 waves; wave = 16M x 32N (two accumulators sharing one
// A fragment).  Weights staged in LDS (ds_load_b128), activations stream from
// global (global_load_b128 + global_prefetch_b8).  Per K-step per wave:
// 2 global b128 + 4 ds b128 -> 2 v_wmma_f32_16x16x32_bf16.
// ---------------------------------------------------------------------------
__global__ __launch_bounds__(256)
void wmma_conv_gemm(const unsigned short* __restrict__ A,
                    const unsigned short* __restrict__ Bt,
                    const float* __restrict__ bias,
                    const float* __restrict__ res,
                    float* __restrict__ C,
                    int M, int N, int K, int act)
{
  __shared__ unsigned short Bs[64 * LDSROW];          // ~37 KB

  const int tid   = (int)threadIdx.x;
  const int wave  = tid >> 5;
  const int lane  = tid & 31;
  const int m0    = (int)blockIdx.x * 64 + (wave & 3) * 16;
  const int nbase = (int)blockIdx.y * 64;
  const int nloc  = (wave >> 2) * 32;                 // 0 or 32 inside block tile
  const int n0    = nbase + nloc;

  const int row = lane & 15;
  const int grp = lane >> 4;                          // K 0-7/16-23 vs 8-15/24-31
  const unsigned short* arow = A + (size_t)(m0 + row) * K;
  const unsigned short* bs0  = Bs + (nloc + row) * LDSROW;
  const unsigned short* bs1  = Bs + (nloc + 16 + row) * LDSROW;

  v8f acc0 = {};
  v8f acc1 = {};

  for (int kc = 0; kc < K; kc += KC) {
    // ---- cooperative stage: Bt[nbase..nbase+64) x [kc..kc+KC) -> LDS ----
    const int QPR = KC / 8;                           // uint4 per row (36)
    for (int f = tid; f < 64 * QPR; f += 256) {
      const int r  = f / QPR;
      const int cq = f - r * QPR;
      uint4 v = {0u, 0u, 0u, 0u};                     // zero-fill rows >= N
      if (nbase + r < N)
        v = *reinterpret_cast<const uint4*>(Bt + (size_t)(nbase + r) * K + kc + cq * 8);
      *reinterpret_cast<uint4*>(&Bs[r * LDSROW + cq * 8]) = v;
    }
    __syncthreads();

    for (int k = 0; k < KC; k += 32) {
      if (kc + k + 64 < K)                            // gfx1250 global_prefetch_b8
        __builtin_prefetch(arow + kc + k + 64, 0, 1);
      BfFrag a, b0, b1;
      a.q[0]  = *reinterpret_cast<const uint4*>(arow + kc + k + 8 * grp);
      a.q[1]  = *reinterpret_cast<const uint4*>(arow + kc + k + 16 + 8 * grp);
      b0.q[0] = *reinterpret_cast<const uint4*>(bs0 + k + 8 * grp);
      b0.q[1] = *reinterpret_cast<const uint4*>(bs0 + k + 16 + 8 * grp);
      b1.q[0] = *reinterpret_cast<const uint4*>(bs1 + k + 8 * grp);
      b1.q[1] = *reinterpret_cast<const uint4*>(bs1 + k + 16 + 8 * grp);
      acc0 = __builtin_amdgcn_wmma_f32_16x16x32_bf16(false, a.v, false, b0.v,
                                                     (short)0, acc0, false, false);
      acc1 = __builtin_amdgcn_wmma_f32_16x16x32_bf16(false, a.v, false, b1.v,
                                                     (short)0, acc1, false, false);
    }
    __syncthreads();
  }

  // ---- fused epilogue: bias + activation (+ residual), fp32 NCHW stores ----
  const int mrow = m0 + grp * 8;
  {
    const int co = n0 + (lane & 15);
    if (co < N && m0 < M) {                           // uniform per wave (N % 16 == 0)
      const float bv = bias ? bias[co] : 0.0f;
      const size_t cb = (size_t)co * M + mrow;
      #pragma unroll
      for (int r = 0; r < 8; ++r) {
        float v = acc0[r] + bv;
        if (act == ACT_RELU)       v = v > 0.0f ? v : 0.0f;
        else if (act == ACT_LEAKY) v = v > 0.0f ? v : 0.1f * v;
        if (res) v += res[cb + r];
        C[cb + r] = v;
      }
    }
  }
  {
    const int co = n0 + 16 + (lane & 15);
    if (co < N && m0 < M) {
      const float bv = bias ? bias[co] : 0.0f;
      const size_t cb = (size_t)co * M + mrow;
      #pragma unroll
      for (int r = 0; r < 8; ++r) {
        float v = acc1[r] + bv;
        if (act == ACT_RELU)       v = v > 0.0f ? v : 0.0f;
        else if (act == ACT_LEAKY) v = v > 0.0f ? v : 0.1f * v;
        if (res) v += res[cb + r];
        C[cb + r] = v;
      }
    }
  }
}

// fp32 NCHW frame -> bf16 im2col [P x Cin*9], zero 'SAME' padding
__global__ void im2col_kernel(const float* __restrict__ X,
                              unsigned short* __restrict__ out, int Cin)
{
  const int p  = (int)blockIdx.x * 256 + (int)threadIdx.x;
  const int ci = (int)blockIdx.y;
  if (p >= PP) return;
  const int y = p / WW, x = p % WW;
  const int K = Cin * 9;
  const size_t base = (size_t)p * K + (size_t)ci * 9;
  const float* Xc = X + (size_t)ci * PP;
  #pragma unroll
  for (int ky = 0; ky < 3; ++ky)
    #pragma unroll
    for (int kx = 0; kx < 3; ++kx) {
      const int yy = y + ky - 1, xx = x + kx - 1;
      const float v = (yy >= 0 && yy < HH && xx >= 0 && xx < WW)
                        ? Xc[yy * WW + xx] : 0.0f;
      out[base + ky * 3 + kx] = f32_to_bf16(v);
    }
}

__global__ void f2bf_kernel(const float* __restrict__ s,
                            unsigned short* __restrict__ d, int n) {
  int i = (int)blockIdx.x * 256 + (int)threadIdx.x;
  if (i < n) d[i] = f32_to_bf16(s[i]);
}
__global__ void zero_f_kernel(float* __restrict__ d, int n) {
  int i = (int)blockIdx.x * 256 + (int)threadIdx.x;
  if (i < n) d[i] = 0.0f;
}
__global__ void zero_i_kernel(int* __restrict__ d, int n) {
  int i = (int)blockIdx.x * 256 + (int)threadIdx.x;
  if (i < n) d[i] = 0;
}
__global__ void copy_f_kernel(const float* __restrict__ s, float* __restrict__ d, int n) {
  int i = (int)blockIdx.x * 256 + (int)threadIdx.x;
  if (i < n) d[i] = s[i];
}
__global__ void copy_i_kernel(const int* __restrict__ s, int* __restrict__ d, int n) {
  int i = (int)blockIdx.x * 256 + (int)threadIdx.x;
  if (i < n) d[i] = s[i];
}

// ---------------- patch-match temporal alignment ----------------
__device__ float pm_cost(const float* __restrict__ cur, const float* __restrict__ last,
                         int p, int x, int y, int ox, int oy)
{
  const int sx = iclampd(x + ox, 0, WW - 1);
  const int sy = iclampd(y + oy, 0, HH - 1);
  const int q = sy * WW + sx;
  float s = 0.0f;
  #pragma unroll
  for (int c = 0; c < 16; ++c) {
    const float d = cur[c * PP + p] - last[c * PP + q];
    s += d * d;
  }
  return s + (float)(ox * ox + oy * oy) * (1.0f / 900.0f);  // sigma=30
}

__global__ void pm_init_kernel(const float* __restrict__ cur, const float* __restrict__ last,
                               const int* __restrict__ off_prev,
                               int* __restrict__ off, float* __restrict__ cost)
{
  int p = (int)blockIdx.x * 256 + (int)threadIdx.x;
  if (p >= PP) return;
  const int x = p % WW, y = p / WW;
  const int ox = off_prev[2 * p], oy = off_prev[2 * p + 1];
  off[2 * p] = ox; off[2 * p + 1] = oy;
  cost[p] = pm_cost(cur, last, p, x, y, ox, oy);
}

__global__ void pm_prop_kernel(const float* __restrict__ cur, const float* __restrict__ last,
                               const int* __restrict__ off_in, const float* __restrict__ c_in,
                               int* __restrict__ off_out, float* __restrict__ c_out,
                               int dy, int dx)
{
  int p = (int)blockIdx.x * 256 + (int)threadIdx.x;
  if (p >= PP) return;
  const int x = p % WW, y = p / WW;
  const int q = iclampd(y + dy, 0, HH - 1) * WW + iclampd(x + dx, 0, WW - 1);
  const int cx = off_in[2 * q], cy = off_in[2 * q + 1];
  const float cc = pm_cost(cur, last, p, x, y, cx, cy);
  const float c0 = c_in[p];
  if (cc < c0) { off_out[2 * p] = cx; off_out[2 * p + 1] = cy; c_out[p] = cc; }
  else { off_out[2 * p] = off_in[2 * p]; off_out[2 * p + 1] = off_in[2 * p + 1]; c_out[p] = c0; }
}

__device__ __forceinline__ unsigned pm_hash(unsigned x) {
  x ^= x >> 16; x *= 0x7feb352dU; x ^= x >> 15; x *= 0x846ca68bU; x ^= x >> 16;
  return x;
}

__global__ void pm_jump_kernel(const float* __restrict__ cur, const float* __restrict__ last,
                               const int* __restrict__ off_in, const float* __restrict__ c_in,
                               int* __restrict__ off_out, float* __restrict__ c_out,
                               unsigned seed)
{
  int p = (int)blockIdx.x * 256 + (int)threadIdx.x;
  if (p >= PP) return;
  const int x = p % WW, y = p / WW;
  const int ox = off_in[2 * p], oy = off_in[2 * p + 1];
  const int cx = ox + (int)(pm_hash(seed ^ (unsigned)(2 * p)) % 5u) - 2;      // [-2,2]
  const int cy = oy + (int)(pm_hash(seed ^ (unsigned)(2 * p + 1)) % 5u) - 2;
  const float cc = pm_cost(cur, last, p, x, y, cx, cy);
  const float c0 = c_in[p];
  if (cc < c0) { off_out[2 * p] = cx; off_out[2 * p + 1] = cy; c_out[p] = cc; }
  else { off_out[2 * p] = ox; off_out[2 * p + 1] = oy; c_out[p] = c0; }
}

// nearest-warp of carried state + exp(-SSE) gating + channel concat (192ch)
__global__ void warp_concat_kernel(const float* __restrict__ hid_l,
                                   const float* __restrict__ hid_h,
                                   const float* __restrict__ last_k,
                                   const float* __restrict__ key_t,
                                   const float* __restrict__ val_t,
                                   const int* __restrict__ off,
                                   float* __restrict__ cat)
{
  int p = (int)blockIdx.x * 256 + (int)threadIdx.x;
  if (p >= PP) return;
  const int x = p % WW, y = p / WW;
  const int q = iclampd(y + off[2 * p + 1], 0, HH - 1) * WW +
                iclampd(x + off[2 * p], 0, WW - 1);
  float s = 0.0f;
  #pragma unroll
  for (int c = 0; c < 16; ++c) {
    const float d = key_t[c * PP + p] - last_k[c * PP + q];
    s += d * d;
  }
  const float wgt = __expf(-s);
  for (int c = 0; c < 64; ++c) cat[(size_t)c * PP + p]          = hid_l[c * PP + q] * wgt;
  for (int c = 0; c < 64; ++c) cat[(size_t)(64 + c) * PP + p]   = hid_h[c * PP + q] * wgt;
  for (int c = 0; c < 64; ++c) cat[(size_t)(128 + c) * PP + p]  = val_t[c * PP + p];
}

// ---------------------------------------------------------------------------
extern "C" void kernel_launch(void* const* d_in, const int* in_sizes, int n_in,
                              void* d_out, int out_size, void* d_ws, size_t ws_size,
                              hipStream_t stream)
{
  (void)in_sizes; (void)n_in; (void)out_size; (void)ws_size;

  const float* x       = (const float*)d_in[0];   // (1,4,64,96,96)
  const float* krb1_b1 = (const float*)d_in[3];
  const float* krb1_b2 = (const float*)d_in[5];
  const float* kconv1_b= (const float*)d_in[7];
  const float* krb2_b1 = (const float*)d_in[9];
  const float* krb2_b2 = (const float*)d_in[11];
  const float* kconv2_b= (const float*)d_in[13];
  const float* val_b   = (const float*)d_in[15];
  const float* cv1_b   = (const float*)d_in[17];
  const float* cv2_b   = (const float*)d_in[19];
  const float* rec_b1  = (const float*)d_in[21];
  const float* rec_b2  = (const float*)d_in[23];

  // ---- bump allocator over d_ws ----
  char* wsbase = (char*)d_ws;
  size_t bump = 0;
  auto alloc = [&](size_t bytes) -> void* {
    size_t o = (bump + 255) & ~(size_t)255;
    bump = o + bytes;
    return (void*)(wsbase + o);
  };

  // bf16 weight copies
  unsigned short* Wkrb1w1 = (unsigned short*)alloc((size_t)64 * 576 * 2);
  unsigned short* Wkrb1w2 = (unsigned short*)alloc((size_t)64 * 576 * 2);
  unsigned short* Wkconv1 = (unsigned short*)alloc((size_t)32 * 576 * 2);
  unsigned short* Wkrb2w1 = (unsigned short*)alloc((size_t)32 * 288 * 2);
  unsigned short* Wkrb2w2 = (unsigned short*)alloc((size_t)32 * 288 * 2);
  unsigned short* Wkconv2 = (unsigned short*)alloc((size_t)16 * 288 * 2);
  unsigned short* Wval    = (unsigned short*)alloc((size_t)64 * 576 * 2);
  unsigned short* Wcv1    = (unsigned short*)alloc((size_t)192 * 1728 * 2);
  unsigned short* Wcv2    = (unsigned short*)alloc((size_t)64 * 1728 * 2);
  unsigned short* Wrec1   = (unsigned short*)alloc((size_t)30 * 64 * 576 * 2);
  unsigned short* Wrec2   = (unsigned short*)alloc((size_t)30 * 64 * 576 * 2);

  unsigned short* imcol = (unsigned short*)alloc((size_t)PP * 1728 * 2);

  float* k64a = (float*)alloc((size_t)64 * PP * 4);
  float* k64b = (float*)alloc((size_t)64 * PP * 4);
  float* k32a = (float*)alloc((size_t)32 * PP * 4);
  float* k32b = (float*)alloc((size_t)32 * PP * 4);
  float* inp_key   = (float*)alloc((size_t)TT * 16 * PP * 4);
  float* inp_value = (float*)alloc((size_t)TT * 64 * PP * 4);
  float* catb = (float*)alloc((size_t)192 * PP * 4);
  float* tmpA = (float*)alloc((size_t)192 * PP * 4);
  float* rb   = (float*)alloc((size_t)64 * PP * 4);
  float* rb2  = (float*)alloc((size_t)64 * PP * 4);

  int*   zoff      = (int*)alloc((size_t)2 * PP * 4);
  int*   ofA       = (int*)alloc((size_t)2 * PP * 4);
  int*   ofB       = (int*)alloc((size_t)2 * PP * 4);
  int*   off_store = (int*)alloc((size_t)TT * 2 * PP * 4);
  float* cA        = (float*)alloc((size_t)PP * 4);
  float* cB        = (float*)alloc((size_t)PP * 4);

  // ---- 1) weights fp32 -> bf16 ----
  struct WC { int idx; unsigned short* dst; int n; };
  const WC wc[11] = {
    {2,  Wkrb1w1, 64 * 576}, {4,  Wkrb1w2, 64 * 576}, {6,  Wkconv1, 32 * 576},
    {8,  Wkrb2w1, 32 * 288}, {10, Wkrb2w2, 32 * 288}, {12, Wkconv2, 16 * 288},
    {14, Wval,    64 * 576}, {16, Wcv1,    192 * 1728}, {18, Wcv2, 64 * 1728},
    {20, Wrec1,   30 * 64 * 576}, {22, Wrec2, 30 * 64 * 576},
  };
  for (int i = 0; i < 11; ++i)
    f2bf_kernel<<<(wc[i].n + 255) / 256, 256, 0, stream>>>(
        (const float*)d_in[wc[i].idx], wc[i].dst, wc[i].n);

  // generic conv: im2col + WMMA GEMM (fused bias/act/residual epilogue)
  auto conv = [&](const float* src, int Cin, const unsigned short* wbf,
                  const float* bias, int Cout, int act, const float* res, float* dst) {
    dim3 gi(PP / 256, Cin);
    im2col_kernel<<<gi, 256, 0, stream>>>(src, imcol, Cin);
    dim3 gg(PP / 64, (Cout + 63) / 64);
    wmma_conv_gemm<<<gg, 256, 0, stream>>>(imcol, wbf, bias, res, dst,
                                           PP, Cout, Cin * 9, act);
  };

  // ---- 2) key / value encoders (per frame) ----
  for (int n = 0; n < TT; ++n) {
    const float* xn = x + (size_t)n * 64 * PP;
    conv(xn,   64, Wkrb1w1, krb1_b1, 64, ACT_RELU, nullptr, k64a);
    conv(k64a, 64, Wkrb1w2, krb1_b2, 64, ACT_NONE, xn,      k64b);   // resblock 1
    conv(k64b, 64, Wkconv1, kconv1_b, 32, ACT_NONE, nullptr, k32a);
    conv(k32a, 32, Wkrb2w1, krb2_b1, 32, ACT_RELU, nullptr, k32b);
    conv(k32b, 32, Wkrb2w2, krb2_b2, 32, ACT_NONE, k32a,    k32b);   // resblock 2 (in-place safe)
    conv(k32b, 32, Wkconv2, kconv2_b, 16, ACT_NONE, nullptr, inp_key + (size_t)n * 16 * PP);
    conv(xn,   64, Wval,    val_b,   64, ACT_NONE, nullptr, inp_value + (size_t)n * 64 * PP);
  }

  // ---- 3) patch-match alignment (8 iters x (4 prop + 1 jump), ping-pong) ----
  zero_i_kernel<<<(2 * PP + 255) / 256, 256, 0, stream>>>(zoff, 2 * PP);
  const int* prev = zoff;
  const int dirs[4][2] = {{0, 1}, {0, -1}, {1, 0}, {-1, 0}};
  for (int t = 0; t < TT; ++t) {
    const float* cur  = inp_key + (size_t)t * 16 * PP;
    const float* last = inp_key + (size_t)(t > 0 ? t - 1 : 0) * 16 * PP;
    pm_init_kernel<<<PP / 256, 256, 0, stream>>>(cur, last, prev, ofA, cA);
    int* oi = ofA; float* ci = cA; int* oo = ofB; float* co = cB;
    for (int it = 0; it < 8; ++it) {
      for (int d = 0; d < 4; ++d) {
        pm_prop_kernel<<<PP / 256, 256, 0, stream>>>(cur, last, oi, ci, oo, co,
                                                     dirs[d][0], dirs[d][1]);
        { int* ti = oi; oi = oo; oo = ti; float* tc = ci; ci = co; co = tc; }
      }
      pm_jump_kernel<<<PP / 256, 256, 0, stream>>>(cur, last, oi, ci, oo, co,
          0x9E3779B9u * (unsigned)(t * 8 + it + 1));
      { int* ti = oi; oi = oo; oo = ti; float* tc = ci; ci = co; co = tc; }
    }
    copy_i_kernel<<<(2 * PP + 255) / 256, 256, 0, stream>>>(oi, off_store + (size_t)t * 2 * PP, 2 * PP);
    prev = off_store + (size_t)t * 2 * PP;
  }

  // ---- 4) recurrent reconstruction ----
  const float* hid_l = nullptr; const float* hid_h = nullptr; const float* last_k = nullptr;
  for (int t = 0; t < TT; ++t) {
    const float* valt = inp_value + (size_t)t * 64 * PP;
    const float* keyt = inp_key + (size_t)t * 16 * PP;
    float* outT = (float*)d_out + (size_t)t * 64 * PP;

    if (t == 0) {
      zero_f_kernel<<<(128 * PP + 255) / 256, 256, 0, stream>>>(catb, 128 * PP);
      copy_f_kernel<<<(64 * PP + 255) / 256, 256, 0, stream>>>(valt, catb + (size_t)128 * PP, 64 * PP);
    } else {
      warp_concat_kernel<<<PP / 256, 256, 0, stream>>>(hid_l, hid_h, last_k, keyt, valt,
                                                       off_store + (size_t)t * 2 * PP, catb);
    }

    // convs_fn: 192 -> 192 (leaky 0.1) -> 64
    conv(catb, 192, Wcv1, cv1_b, 192, ACT_LEAKY, nullptr, tmpA);
    conv(tmpA, 192, Wcv2, cv2_b, 64,  ACT_NONE,  nullptr, rb);

    // recon_fn: 30 resblocks of 64ch 3x3 convs (the compute core -> WMMA)
    for (int i = 0; i < 30; ++i) {
      conv(rb,  64, Wrec1 + (size_t)i * 64 * 576, rec_b1 + i * 64, 64, ACT_RELU, nullptr, rb2);
      float* dst = (i == 29) ? outT : rb;
      conv(rb2, 64, Wrec2 + (size_t)i * 64 * 576, rec_b2 + i * 64, 64, ACT_NONE, rb, dst);
    }

    hid_l = valt; hid_h = outT; last_k = keyt;
  }
}
